// AttentionModel_64055142252851
// MI455X (gfx1250) — matrix operand order repaired and checked by reading, not compiled, not run on hardware
//
#include <hip/hip_runtime.h>
#include <hip/hip_bf16.h>

typedef unsigned short u16;
typedef unsigned int   u32;
typedef unsigned long long u64;

typedef __attribute__((ext_vector_type(16))) __bf16 v16bf;
typedef __attribute__((ext_vector_type(8)))  float  v8f;
typedef __attribute__((ext_vector_type(8)))  unsigned short us8;
typedef __attribute__((ext_vector_type(4)))  unsigned int u32x4;
typedef __attribute__((ext_vector_type(8)))  int i32x8;
typedef __attribute__((ext_vector_type(4)))  int i32x4;

#define B_   16
#define T_   1024
#define F_   257
#define H_   512
#define M_   (B_ * T_)      // 16384 rows
#define KXP  288            // 257 padded to 32
#define NMP  384            // 257 padded to 128 (mask N)
#define ATTK 160            // 144-wide band padded to 32-multiple
#define KTLD 1184           // 128 zero head + 1024 + 32 zero tail

union Frag { v16bf v; us8 h[2]; };

__device__ __forceinline__ u16 f2bf(float f) {
  u32 u = __float_as_uint(f);
  u32 r = (u + 0x7FFFu + ((u >> 16) & 1u)) >> 16;   // round-to-nearest-even
  return (u16)r;
}
__device__ __forceinline__ float sigm(float x) { return 1.0f / (1.0f + __expf(-x)); }
__device__ __forceinline__ v8f vzero() { v8f z = {0.f,0.f,0.f,0.f,0.f,0.f,0.f,0.f}; return z; }

// A fragment: 16(M) x 32(K) bf16, row-major K-contiguous source.
// Lane l<16 holds row M=l, K in {kb..kb+15}, kb = 0 (lo) / 16 (hi).
__device__ __forceinline__ v16bf frag_A(const u16* A, size_t lda, int row0, int k0, int lane) {
  const u16* p = A + (size_t)(row0 + (lane & 15)) * lda + (size_t)(k0 + ((lane >> 4) << 4));
  Frag f; f.h[0] = *(const us8*)p; f.h[1] = *(const us8*)(p + 8);
  return f.v;
}
// B fragment: 32(K) x 16(N) from Bt stored (N x K) row-major: lane = column N,
// lanes 0-15 hold K=k0..k0+15, lanes 16-31 hold K=k0+16..k0+31 (contiguous 16).
__device__ __forceinline__ v16bf frag_B(const u16* Bt, size_t ldb, int col0, int k0, int lane) {
  const u16* p = Bt + (size_t)(col0 + (lane & 15)) * ldb + (size_t)(k0 + ((lane >> 4) << 4));
  Frag f; f.h[0] = *(const us8*)p; f.h[1] = *(const us8*)(p + 8);
  return f.v;
}
__device__ __forceinline__ v8f wmma_bf16(v16bf a, v16bf b, v8f c) {
  return __builtin_amdgcn_wmma_f32_16x16x32_bf16(false, a, false, b, (short)0, c, false, false);
}

// ---------------- fills / converts ----------------
__global__ void fill_zero_f32(float* p, size_t n) {
  for (size_t i = blockIdx.x * (size_t)blockDim.x + threadIdx.x; i < n;
       i += (size_t)gridDim.x * blockDim.x) p[i] = 0.0f;
}
__global__ void fill_zero_u16(u16* p, size_t n) {
  for (size_t i = blockIdx.x * (size_t)blockDim.x + threadIdx.x; i < n;
       i += (size_t)gridDim.x * blockDim.x) p[i] = 0;
}
__global__ void init_state(u16* hbuf, u32* syncc) {
  int i = blockIdx.x * blockDim.x + threadIdx.x;
  if (i < 2 * 2 * 16 * 512) hbuf[i] = 0;
  if (i < 2) syncc[i] = 0;
}
__global__ void cvt_bf16_pad(const float* __restrict__ src, u16* __restrict__ dst,
                             int srows, int scols, int drows, int dcols) {
  size_t i = blockIdx.x * (size_t)blockDim.x + threadIdx.x;
  size_t total = (size_t)drows * dcols;
  if (i >= total) return;
  int r = (int)(i / dcols), c = (int)(i % dcols);
  float v = (r < srows && c < scols) ? src[(size_t)r * scols + c] : 0.0f;
  dst[i] = f2bf(v);
}
__global__ void pad_f32(const float* __restrict__ s, float* __restrict__ d, int n, int np) {
  int i = blockIdx.x * blockDim.x + threadIdx.x;
  if (i < np) d[i] = (i < n) ? s[i] : 0.0f;
}

// ---------------- generic WMMA GEMM: C(MxN) = A(MxK) * Bt(NxK)^T (+bias, epilogue) ----------
// EP: 0 = f32 store, 1 = bf16 store, 2 = tanh->bf16, 3 = sigmoid(v)*Xin -> f32
template <int EP>
__global__ __launch_bounds__(128) void gemm_bf16_kernel(
    const u16* __restrict__ A, size_t lda,
    const u16* __restrict__ Bt, size_t ldb,
    const float* __restrict__ bias,
    void* __restrict__ Cv, size_t ldc,
    int Nreal, int K,
    const float* __restrict__ Xin, size_t ldx)
{
  const int lane = threadIdx.x & 31;
  const int wv   = threadIdx.x >> 5;               // 4 waves: 2x2 of 16x64 tiles
  const int m0   = blockIdx.y * 32 + (wv >> 1) * 16;
  const int nb   = blockIdx.x * 128 + (wv & 1) * 64;
  v8f acc[4] = { vzero(), vzero(), vzero(), vzero() };
  for (int k0 = 0; k0 < K; k0 += 32) {
    v16bf a = frag_A(A, lda, m0, k0, lane);
#pragma unroll
    for (int j = 0; j < 4; ++j) {
      v16bf b = frag_B(Bt, ldb, nb + j * 16, k0, lane);
      acc[j] = wmma_bf16(a, b, acc[j]);
    }
    __builtin_prefetch(A + (size_t)(m0 + (lane & 15)) * lda + k0 + 128, 0, 0);
  }
  const int hi = lane >> 4;
#pragma unroll
  for (int j = 0; j < 4; ++j) {
    int n = nb + j * 16 + (lane & 15);
    if (n >= Nreal) continue;
    float bv = bias ? bias[n] : 0.0f;
#pragma unroll
    for (int i = 0; i < 8; ++i) {
      int m = m0 + i + 8 * hi;
      float v = acc[j][i] + bv;
      size_t idx = (size_t)m * ldc + n;
      if (EP == 0)      ((float*)Cv)[idx] = v;
      else if (EP == 1) ((u16*)Cv)[idx] = f2bf(v);
      else if (EP == 2) ((u16*)Cv)[idx] = f2bf(tanhf(v));
      else              ((float*)Cv)[idx] = Xin[(size_t)m * ldx + n] * sigm(v);
    }
  }
}

// ---------------- k transpose: (B,T,H) -> (B,H,KTLD) at column s+128 ----------------
__global__ __launch_bounds__(256) void transpose_k_kernel(const u16* __restrict__ k,
                                                          u16* __restrict__ kT) {
  __shared__ u16 tile[32][40];
  int b = blockIdx.z, s0 = blockIdx.x * 32, h0 = blockIdx.y * 32;
  int x = threadIdx.x & 31, y = threadIdx.x >> 5;
  for (int yy = y; yy < 32; yy += 8)
    tile[yy][x] = k[((size_t)b * T_ + s0 + yy) * H_ + h0 + x];
  __syncthreads();
  for (int yy = y; yy < 32; yy += 8)
    kT[((size_t)b * H_ + h0 + yy) * KTLD + 128 + s0 + x] = tile[x][yy];
}

// ---------------- persistent dual-LSTM with LDS-resident w_hh (TDM loaded) ----------------
__global__ __launch_bounds__(256) void lstm_kernel(
    const float* __restrict__ xg0, const float* __restrict__ xg1,
    const u16* __restrict__ whh0, const u16* __restrict__ whh1,
    const float* __restrict__ bhh0, const float* __restrict__ bhh1,
    u16* __restrict__ hbuf, u32* __restrict__ syncc,
    u16* __restrict__ out0, u16* __restrict__ out1)
{
  extern __shared__ char smem_raw[];
  u16*   whh_s = (u16*)smem_raw;                      // 128 x 512 bf16 = 128KB
  float* gbuf  = (float*)(smem_raw + 128 * 512 * 2);  // 128 cols x 16 batch = 8KB
  float* cst   = gbuf + 128 * 16;                     // 16 x 32 cell state = 2KB

  const int lstm  = blockIdx.x >> 4;       // 0: k-LSTM, 1: q-LSTM
  const int part  = blockIdx.x & 15;       // 16 parts x 32 h-dims
  const int hbase = part * 32;
  const float* xg  = lstm ? xg1 : xg0;
  const u16*   whh = lstm ? whh1 : whh0;
  const float* bhh = lstm ? bhh1 : bhh0;
  u16* out = lstm ? out1 : out0;
  const size_t oldc = lstm ? 1024 : 512;
  const int    ooff = lstm ? 512 : 0;
  const int lane = threadIdx.x & 31;
  const int wv   = threadIdx.x >> 5;       // 8 waves -> 8 n-tiles of 16 gate cols

  // Load this block's 4x(32x512) w_hh slices (gates i,f,g,o for hbase..hbase+31) into LDS.
#if __has_builtin(__builtin_amdgcn_tensor_load_to_lds)
  if (threadIdx.x < 32) {
    for (int g = 0; g < 4; ++g) {
      u64 gaddr = (u64)(const void*)(whh + (size_t)(g * 512 + hbase) * 512);
      u32 laddr = (u32)(u64)(void*)(whh_s + g * 32 * 512);
      u32x4 d0;
      d0[0] = 1u;                                            // count=1, user descriptor
      d0[1] = laddr;                                         // lds_addr
      d0[2] = (u32)(gaddr & 0xFFFFFFFFu);                    // global_addr lo
      d0[3] = (u32)((gaddr >> 32) & 0x1FFFFFFu) | (2u << 30);// global_addr hi | type=2
      i32x8 d1;
      d1[0] = (int)(1u << 16);            // data_size=1 (2 bytes/elem)
      d1[1] = (int)(512u << 16);          // tensor_dim0 = 512 (lo16 at bit48)
      d1[2] = (int)(2048u << 16);         // dim0 hi16=0 | tensor_dim1 lo16 = 2048
      d1[3] = (int)(512u << 16);          // dim1 hi16=0 | tile_dim0 = 512
      d1[4] = 32;                         // tile_dim1 = 32 rows
      d1[5] = 512;                        // tensor_dim0_stride = 512 elems
      d1[6] = 0;
      d1[7] = 0;
      i32x4 dz4 = {0, 0, 0, 0};
      i32x8 dz8 = {0, 0, 0, 0, 0, 0, 0, 0};
      __builtin_amdgcn_tensor_load_to_lds(d0, d1, dz4, dz4, dz8, 0);
    }
    __builtin_amdgcn_s_wait_tensorcnt(0);
  }
#else
  for (int e = threadIdx.x * 8; e < 128 * 512; e += 256 * 8) {
    int l = e >> 9, c = e & 511;
    int grow = ((l >> 5) << 9) + hbase + (l & 31);
    *(us8*)(whh_s + e) = *(const us8*)(whh + (size_t)grow * 512 + c);
  }
#endif
  for (int i = threadIdx.x; i < 512; i += 256) cst[i] = 0.0f;
  __syncthreads();

  u16* hA = hbuf + (size_t)lstm * 2 * 16 * 512;   // phase 0 (zero-initialized)
  u16* hB = hA + 16 * 512;                        // phase 1

  for (int t = 0; t < T_; ++t) {
    const u16* hin = (t & 1) ? hB : hA;
    u16* hout = (t & 1) ? hA : hB;
    // G(16 x 16cols) = h_prev(16x512) @ whh_slice^T for this wave's 16 gate cols
    v8f acc = vzero();
#pragma unroll 4
    for (int k0 = 0; k0 < 512; k0 += 32) {
      v16bf a = frag_A(hin, 512, 0, k0, lane);            // global bf16 h state
      v16bf b = frag_B(whh_s, 512, wv * 16, k0, lane);    // LDS weights (ds reads)
      acc = wmma_bf16(a, b, acc);
    }
    {
      int l = wv * 16 + (lane & 15);                       // local gate col 0..127
      int gcol = ((l >> 5) << 9) + hbase + (l & 31);       // global gate col
      int hi2 = lane >> 4;
      float bb = bhh[gcol];
#pragma unroll
      for (int i = 0; i < 8; ++i) {
        int bidx = i + 8 * hi2;                            // batch row
        gbuf[l * 16 + bidx] = acc[i] + xg[((size_t)bidx * T_ + t) * 2048 + gcol] + bb;
      }
    }
    __syncthreads();
    for (int idx = threadIdx.x; idx < 512; idx += 256) {   // (batch, jl) pairs
      int bidx = idx >> 5, jl = idx & 31;
      float gi = gbuf[(jl) * 16 + bidx];
      float gf = gbuf[(32 + jl) * 16 + bidx];
      float gg = gbuf[(64 + jl) * 16 + bidx];
      float go = gbuf[(96 + jl) * 16 + bidx];
      float cn = sigm(gf) * cst[idx] + sigm(gi) * tanhf(gg);
      cst[idx] = cn;
      u16 hb = f2bf(sigm(go) * tanhf(cn));
      int hdim = hbase + jl;
      hout[bidx * 512 + hdim] = hb;
      out[((size_t)bidx * T_ + t) * oldc + ooff + hdim] = hb;
    }
    __threadfence();
    __syncthreads();
    if (threadIdx.x == 0) {                                 // 16-block producer/consumer sync
      atomicAdd(&syncc[lstm], 1u);
      u32 target = 16u * (u32)(t + 1);
      while (__hip_atomic_load(&syncc[lstm], __ATOMIC_RELAXED, __HIP_MEMORY_SCOPE_AGENT) < target)
        __builtin_amdgcn_s_sleep(1);
    }
    __builtin_amdgcn_s_cluster_barrier();   // NOP outside cluster dispatch
    __syncthreads();
  }
}

// ---------------- fused banded attention: scores -> exp/band -> normalize -> c ------------
__global__ __launch_bounds__(128) void attn_kernel(
    const u16* __restrict__ qs, const u16* __restrict__ kbf,
    const u16* __restrict__ kT, float* __restrict__ attn_out,
    u16* __restrict__ cq)
{
  __shared__ float esc[16 * ATTK];
  __shared__ float rowsum[16];
  __shared__ u16 attn_s[16 * ATTK];
  const int t0 = blockIdx.x * 16;
  const int b  = blockIdx.y;
  const int lane = threadIdx.x & 31;
  const int wv   = threadIdx.x >> 5;
  const int hi   = lane >> 4;
  for (int i = threadIdx.x; i < 16 * ATTK; i += 128) { esc[i] = 0.0f; attn_s[i] = 0; }
  if (threadIdx.x < 16) rowsum[threadIdx.x] = 0.0f;
  __syncthreads();

  const u16* kb = kbf + (size_t)b * T_ * H_;
  for (int st = wv; st < 9; st += 4) {              // 9 s-tiles cover band [t0-128, t0+15]
    int s0 = t0 - 128 + st * 16;
    if (s0 < 0) continue;
    v8f acc = vzero();
    for (int k0 = 0; k0 < H_; k0 += 32) {
      v16bf a = frag_A(qs, H_, t0, k0, lane);
      v16bf bb = frag_B(kb, H_, s0, k0, lane);
      acc = wmma_bf16(a, bb, acc);
    }
    int s = s0 + (lane & 15);
#pragma unroll
    for (int i = 0; i < 8; ++i) {
      int tr = t0 + i + 8 * hi;
      float e = (s <= tr && s >= tr - 128) ? __expf(acc[i]) : 0.0f;
      esc[(tr - t0) * ATTK + st * 16 + (lane & 15)] = e;
      atomicAdd(&rowsum[tr - t0], e);               // ds_add_f32
    }
  }
  __syncthreads();
  for (int i = threadIdx.x; i < 16 * ATTK; i += 128) {
    int r = i / ATTK, cc = i % ATTK;
    float a = esc[i] / (rowsum[r] + 1e-10f);
    attn_s[i] = f2bf(a);
    int s = t0 - 128 + cc;
    if (cc < 144 && s >= 0)
      attn_out[((size_t)b * T_ + t0 + r) * T_ + s] = a;
  }
  __syncthreads();
  // c(16 x 512) = attn_band(16 x 160) @ kT rows (K = s dim, zero-padded head/tail)
  const u16* kTb = kT + (size_t)b * H_ * KTLD;
  for (int jj = 0; jj < 8; ++jj) {
    int n0 = (wv * 8 + jj) * 16;
    v8f acc = vzero();
#pragma unroll
    for (int k0 = 0; k0 < ATTK; k0 += 32) {
      v16bf a = frag_A(attn_s, ATTK, 0, k0, lane);  // from LDS
      const u16* p = kTb + (size_t)(n0 + (lane & 15)) * KTLD + t0 + k0 + ((lane >> 4) << 4);
      Frag fb; fb.h[0] = *(const us8*)p; fb.h[1] = *(const us8*)(p + 8);
      acc = wmma_bf16(a, fb.v, acc);
    }
#pragma unroll
    for (int i = 0; i < 8; ++i) {
      int tr = t0 + i + 8 * hi;
      cq[((size_t)b * T_ + tr) * 1024 + n0 + (lane & 15)] = f2bf(acc[i]);
    }
  }
}

// ---------------- host ----------------
static inline size_t alignup(size_t x) { return (x + 255) & ~(size_t)255; }

extern "C" void kernel_launch(void* const* d_in, const int* in_sizes, int n_in,
                              void* d_out, int out_size, void* d_ws, size_t ws_size,
                              hipStream_t stream) {
  const float* x       = (const float*)d_in[0];
  const float* feat_w  = (const float*)d_in[1];
  const float* feat_b  = (const float*)d_in[2];
  const float* k_w_ih  = (const float*)d_in[3];
  const float* k_w_hh  = (const float*)d_in[4];
  const float* k_b_ih  = (const float*)d_in[5];
  const float* k_b_hh  = (const float*)d_in[6];
  const float* q_w_ih  = (const float*)d_in[7];
  const float* q_w_hh  = (const float*)d_in[8];
  const float* q_b_ih  = (const float*)d_in[9];
  const float* q_b_hh  = (const float*)d_in[10];
  const float* score_w = (const float*)d_in[11];
  const float* enh_w   = (const float*)d_in[12];
  const float* enh_b   = (const float*)d_in[13];
  const float* mask_w  = (const float*)d_in[14];
  const float* mask_b  = (const float*)d_in[15];
  (void)in_sizes; (void)n_in; (void)out_size; (void)ws_size;

  char* w = (char*)d_ws;
  auto take = [&](size_t bytes) { char* p = w; w += alignup(bytes); return p; };
  u16* x_bf     = (u16*)take((size_t)M_ * KXP * 2);
  u16* featw_bf = (u16*)take((size_t)H_ * KXP * 2);
  u16* xf_bf    = (u16*)take((size_t)M_ * H_ * 2);
  u16* kwih_bf  = (u16*)take((size_t)2048 * 512 * 2);
  u16* qwih_bf  = (u16*)take((size_t)2048 * 512 * 2);
  u16* kwhh_bf  = (u16*)take((size_t)2048 * 512 * 2);
  u16* qwhh_bf  = (u16*)take((size_t)2048 * 512 * 2);
  u16* scorew_bf= (u16*)take((size_t)512 * 512 * 2);
  u16* enhw_bf  = (u16*)take((size_t)512 * 1024 * 2);
  u16* maskw_bf = (u16*)take((size_t)NMP * 512 * 2);
  float* maskb_p= (float*)take((size_t)NMP * 4);
  float* xg_k   = (float*)take((size_t)M_ * 2048 * 4);
  float* xg_q   = (float*)take((size_t)M_ * 2048 * 4);
  u16* k_bf     = (u16*)take((size_t)M_ * 512 * 2);
  u16* cq_bf    = (u16*)take((size_t)M_ * 1024 * 2);
  u16* qs_bf    = (u16*)take((size_t)M_ * 512 * 2);
  u16* kT_bf    = (u16*)take((size_t)B_ * 512 * KTLD * 2);
  u16* enh_bf   = (u16*)take((size_t)M_ * 512 * 2);
  u16* hbuf     = (u16*)take((size_t)2 * 2 * 16 * 512 * 2);
  u32* syncc    = (u32*)take(256);

  float* out0 = (float*)d_out;                                // (B,T,F) = x * mask
  float* attn_out = out0 + (size_t)M_ * F_;                   // (B,T,T)

  // state / zero fills (every call: ws & d_out are poisoned once before timing)
  init_state<<<128, 256, 0, stream>>>(hbuf, syncc);
  fill_zero_f32<<<4096, 256, 0, stream>>>(attn_out, (size_t)B_ * T_ * T_);
  fill_zero_u16<<<4096, 256, 0, stream>>>(kT_bf, (size_t)B_ * 512 * KTLD);

  // bf16 conversions (weights are (N,K) row-major already; pad K/N where needed)
  auto cvt = [&](const float* s, u16* d, int sr, int sc, int dr, int dc) {
    size_t total = (size_t)dr * dc;
    cvt_bf16_pad<<<(unsigned)((total + 255) / 256), 256, 0, stream>>>(s, d, sr, sc, dr, dc);
  };
  cvt(x,       x_bf,     M_,  F_,  M_,  KXP);
  cvt(feat_w,  featw_bf, H_,  F_,  H_,  KXP);
  cvt(k_w_ih,  kwih_bf,  2048, 512, 2048, 512);
  cvt(q_w_ih,  qwih_bf,  2048, 512, 2048, 512);
  cvt(k_w_hh,  kwhh_bf,  2048, 512, 2048, 512);
  cvt(q_w_hh,  qwhh_bf,  2048, 512, 2048, 512);
  cvt(score_w, scorew_bf, 512, 512, 512, 512);
  cvt(enh_w,   enhw_bf,  512, 1024, 512, 1024);
  cvt(mask_w,  maskw_bf, F_,  512, NMP, 512);
  pad_f32<<<2, 256, 0, stream>>>(mask_b, maskb_p, F_, NMP);

  const dim3 blk(128);
  // xf = x @ feat_w.T + feat_b  -> bf16
  gemm_bf16_kernel<1><<<dim3(512 / 128, M_ / 32), blk, 0, stream>>>(
      x_bf, KXP, featw_bf, KXP, feat_b, xf_bf, 512, 512, KXP, nullptr, 0);
  // xg = xf @ w_ih.T + b_ih  -> f32 (both LSTMs)
  gemm_bf16_kernel<0><<<dim3(2048 / 128, M_ / 32), blk, 0, stream>>>(
      xf_bf, 512, kwih_bf, 512, k_b_ih, xg_k, 2048, 2048, 512, nullptr, 0);
  gemm_bf16_kernel<0><<<dim3(2048 / 128, M_ / 32), blk, 0, stream>>>(
      xf_bf, 512, qwih_bf, 512, q_b_ih, xg_q, 2048, 2048, 512, nullptr, 0);

  // persistent dual-LSTM: 32 blocks (2 LSTMs x 16 h-partitions)
  const int lstm_smem = 128 * 512 * 2 + 128 * 16 * 4 + 512 * 4;
  (void)hipFuncSetAttribute((const void*)lstm_kernel,
                            hipFuncAttributeMaxDynamicSharedMemorySize, lstm_smem);
  lstm_kernel<<<32, 256, lstm_smem, stream>>>(
      xg_k, xg_q, kwhh_bf, qwhh_bf, k_b_hh, q_b_hh, hbuf, syncc,
      k_bf /*k out, ld 512*/, cq_bf /*q out at col 512, ld 1024*/);

  // qs = q @ score_w.T  (A = cq columns 512..1023)
  gemm_bf16_kernel<1><<<dim3(512 / 128, M_ / 32), blk, 0, stream>>>(
      cq_bf + 512, 1024, scorew_bf, 512, nullptr, qs_bf, 512, 512, 512, nullptr, 0);

  // k transpose for the attn @ k GEMM
  transpose_k_kernel<<<dim3(T_ / 32, H_ / 32, B_), 256, 0, stream>>>(k_bf, kT_bf);

  // fused banded attention: writes attn (f32, d_out) and c (bf16, cq cols 0..511)
  attn_kernel<<<dim3(T_ / 16, B_), 128, 0, stream>>>(qs_bf, k_bf, kT_bf, attn_out, cq_bf);

  // enh = tanh([c,q] @ enh_w.T + enh_b) -> bf16
  gemm_bf16_kernel<2><<<dim3(512 / 128, M_ / 32), blk, 0, stream>>>(
      cq_bf, 1024, enhw_bf, 1024, enh_b, enh_bf, 512, 512, 1024, nullptr, 0);
  // out = x * sigmoid(enh @ mask_w.T + mask_b) -> f32 (N guarded at 257)
  gemm_bf16_kernel<3><<<dim3(NMP / 128, M_ / 32), blk, 0, stream>>>(
      enh_bf, 512, maskw_bf, 512, maskb_p, out0, F_, F_, 512, x, F_);
}
